// LuGTP_14413910246019
// MI455X (gfx1250) — compile-verified, matrix operand order checked by reading, not run on hardware
//
#include <hip/hip_runtime.h>

#define B_   64
#define N_   512
#define FIN_ 1024
#define H_   128
#define L_   512
#define K1_  256
#define K2_  128
#define K3_  64

typedef __attribute__((ext_vector_type(8)))  float          v8f;
typedef __attribute__((ext_vector_type(16))) __bf16         v16bf;
typedef __attribute__((ext_vector_type(4)))  float          f32x4;
typedef __attribute__((ext_vector_type(4)))  unsigned int   u32x4;

union BF16x16 {
  v16bf v;
  unsigned short u[16];
  u32x4 q[2];
};

union BF16x8 {
  u32x4 q;
  unsigned short u[8];
};

__device__ __forceinline__ unsigned short f2bf(float f) {
  unsigned int u = __float_as_uint(f);
  u += 0x7FFFu + ((u >> 16) & 1u);           // round-to-nearest-even
  return (unsigned short)(u >> 16);
}

__device__ __forceinline__ float bf2f(unsigned short s) {
  unsigned int u = ((unsigned int)s) << 16;
  return __uint_as_float(u);
}

// ---------------------------------------------------------------------------
// Generic batched GEMM: out[z] = epilogue( A[z] @ B[z] )
//   A: row-major [M x K], lda == K; f32 (ABF16=0) or bf16 (ABF16=1)
//   Bt: bf16 packed transposed [Np x K], Np = roundup(N,128), zero padded
//   Output: Cf f32 row-major [M x N]  OR  CT bf16 transposed [N x M]
//   epilogue: optional rowscale[z*M+m], per-col bias, relu (applied pre-convert)
// One wave (32 threads) computes a 16x128 strip via 8 x V_WMMA_F32_16X16X32_BF16.
// grid = (ceil(N/128), M/16, Z)
// ---------------------------------------------------------------------------
template <int ABF16>
__global__ void gemm_wmma(const void* __restrict__ Aop,
                          const unsigned short* __restrict__ Bt,
                          float* __restrict__ Cf,
                          unsigned short* __restrict__ CT,
                          int M, int N, int K,
                          long long sA, long long sB, long long sC,
                          const float* __restrict__ rowscale,
                          const float* __restrict__ bias,
                          int relu) {
  const int lane = threadIdx.x & 31;
  const int z    = blockIdx.z;
  const int row0 = blockIdx.y * 16;
  const int col0 = blockIdx.x * 128;
  const int r    = lane & 15;
  const int kb   = (lane >> 4) << 3;   // 0 or 8 (CDNA5 16-bit fragment K split)
  const int rhi  = (lane >> 4) << 3;

  const unsigned short* Bb = Bt + (long long)z * sB;

  v8f acc[8] = {};

  for (int k0 = 0; k0 < K; k0 += 32) {
    // ---- A fragment: row (row0+r); elems 0..7 = K k0+kb.., elems 8..15 = K k0+16+kb..
    BF16x16 af;
    if (ABF16) {
      const unsigned short* ap =
          (const unsigned short*)Aop + (long long)z * sA + (long long)(row0 + r) * K + k0 + kb;
      af.q[0] = *(const u32x4*)(ap);
      af.q[1] = *(const u32x4*)(ap + 16);
    } else {
      const float* ap =
          (const float*)Aop + (long long)z * sA + (long long)(row0 + r) * K + k0 + kb;
      f32x4 a0 = *(const f32x4*)(ap);
      f32x4 a1 = *(const f32x4*)(ap + 4);
      f32x4 a2 = *(const f32x4*)(ap + 16);
      f32x4 a3 = *(const f32x4*)(ap + 20);
      af.u[0]  = f2bf(a0.x); af.u[1]  = f2bf(a0.y); af.u[2]  = f2bf(a0.z); af.u[3]  = f2bf(a0.w);
      af.u[4]  = f2bf(a1.x); af.u[5]  = f2bf(a1.y); af.u[6]  = f2bf(a1.z); af.u[7]  = f2bf(a1.w);
      af.u[8]  = f2bf(a2.x); af.u[9]  = f2bf(a2.y); af.u[10] = f2bf(a2.z); af.u[11] = f2bf(a2.w);
      af.u[12] = f2bf(a3.x); af.u[13] = f2bf(a3.y); af.u[14] = f2bf(a3.z); af.u[15] = f2bf(a3.w);
    }
#pragma unroll
    for (int j = 0; j < 8; ++j) {
      const unsigned short* bp = Bb + (long long)(col0 + 16 * j + r) * K + k0 + kb;
      BF16x16 bf;
      bf.q[0] = *(const u32x4*)(bp);        // K = k0+kb .. +7
      bf.q[1] = *(const u32x4*)(bp + 16);   // K = k0+16+kb .. +7
      acc[j] = __builtin_amdgcn_wmma_f32_16x16x32_bf16(
          false, af.v, false, bf.v, (short)0, acc[j], false, false);
    }
  }

  // ---- epilogue + store: lane holds column (col0+16j+r); VGPR i -> row row0+i+rhi
#pragma unroll
  for (int j = 0; j < 8; ++j) {
    int c = col0 + 16 * j + r;
    if (c < N) {
      float vv[8];
#pragma unroll
      for (int i = 0; i < 8; ++i) {
        int m = row0 + i + rhi;
        float v = acc[j][i];
        if (rowscale) v *= rowscale[(long long)z * M + m];
        if (bias)     v += bias[c];
        if (relu)     v = fmaxf(v, 0.0f);
        vv[i] = v;
      }
      if (CT) {  // bf16 transposed [N x M]: 8 contiguous rows -> one 16B store
        BF16x8 st;
#pragma unroll
        for (int i = 0; i < 8; ++i) st.u[i] = f2bf(vv[i]);
        *(u32x4*)(CT + (long long)z * sC + (long long)c * M + row0 + rhi) = st.q;
      } else {   // f32 row-major [M x N]
#pragma unroll
        for (int i = 0; i < 8; ++i)
          Cf[(long long)z * sC + (long long)(row0 + i + rhi) * N + c] = vv[i];
      }
    }
  }
}

// ---------------------------------------------------------------------------
// pack: dst[z][c][r] = bf16(src[z][r][c]) with column padding to Cp (zeros)
// ---------------------------------------------------------------------------
__global__ void packT_kernel(const float* __restrict__ src, unsigned short* __restrict__ dst,
                             int R, int C, int Cp, long long total) {
  long long i = (long long)blockIdx.x * 256 + threadIdx.x;
  if (i >= total) return;
  int r = (int)(i % R);
  long long t = i / R;
  int c = (int)(t % Cp);
  long long z = t / Cp;
  float v = (c < C) ? src[z * (long long)R * C + (long long)r * C + c] : 0.0f;
  dst[i] = f2bf(v);
}

// conv weights -> bf16 [768 x 1024]; col = base + co*kw + j ; value = w[co, ci, j]
__global__ void pack_convw_kernel(const float* __restrict__ cw0, const float* __restrict__ cw1,
                                  const float* __restrict__ cw2, const float* __restrict__ cw3,
                                  unsigned short* __restrict__ dst) {
  int i = blockIdx.x * 256 + threadIdx.x;
  if (i >= 768 * 1024) return;
  int col = i / 1024, ci = i % 1024;
  const float* w; int kw, base;
  if (col < 96)       { w = cw0; kw = 3; base = 0;   }
  else if (col < 256) { w = cw1; kw = 5; base = 96;  }
  else if (col < 480) { w = cw2; kw = 7; base = 256; }
  else                { w = cw3; kw = 9; base = 480; }
  int local = col - base;
  int co = local / kw, j = local % kw;
  dst[i] = f2bf(w[((long long)co * FIN_ + ci) * kw + j]);
}

// plain f32 -> bf16 elementwise (row-major A operand pre-conversion)
__global__ void cvt_bf16_kernel(const float* __restrict__ src, unsigned short* __restrict__ dst,
                                long long n) {
  long long i = (long long)blockIdx.x * 256 + threadIdx.x;
  if (i < n) dst[i] = f2bf(src[i]);
}

__global__ void fill_kernel(float* __restrict__ p, float v, int n) {
  int i = blockIdx.x * 256 + threadIdx.x;
  if (i < n) p[i] = v;
}

// d[b*n+r] = deg>0 ? rsqrt(deg) : 0
__global__ void deg_kernel(const float* __restrict__ A, float* __restrict__ d, int n, int total) {
  int i = blockIdx.x * 256 + threadIdx.x;
  if (i >= total) return;
  int b = i / n, r = i % n;
  const float* ar = A + (long long)b * n * n + (long long)r * n;
  float s = 0.0f;
  for (int c = 0; c < n; ++c) s += ar[c];
  d[i] = (s > 0.0f) ? rsqrtf(s) : 0.0f;
}

// score = tanh( (X@p) / (||p|| + 1e-12) ), X rows of length H
__global__ void score_kernel(const float* __restrict__ X, const float* __restrict__ p,
                             float* __restrict__ score, int total) {
  int i = blockIdx.x * 256 + threadIdx.x;
  if (i >= total) return;
  const float* x = X + (long long)i * H_;
  float dot = 0.0f, pn = 0.0f;
  for (int h = 0; h < H_; ++h) { dot += x[h] * p[h]; pn += p[h] * p[h]; }
  score[i] = tanhf(dot / (sqrtf(pn) + 1e-12f));
}

// one block per batch: iterative argmax top-k (order-invariant downstream)
__global__ void topk_kernel(const float* __restrict__ score, int* __restrict__ idx, int n, int k) {
  __shared__ float sv[512];
  __shared__ float rv[256];
  __shared__ int   ri[256];
  int b = blockIdx.x, tid = threadIdx.x;
  for (int i = tid; i < n; i += 256) sv[i] = score[(long long)b * n + i];
  __syncthreads();
  for (int t = 0; t < k; ++t) {
    float best = -3.4e38f; int bi = -1;
    for (int i = tid; i < n; i += 256) {
      float v = sv[i];
      if (v > best || (v == best && bi >= 0 && i < bi)) { best = v; bi = i; }
    }
    rv[tid] = best; ri[tid] = bi;
    __syncthreads();
    for (int s = 128; s > 0; s >>= 1) {
      if (tid < s) {
        bool take = (ri[tid] < 0) ||
                    (ri[tid + s] >= 0 && (rv[tid + s] > rv[tid] ||
                     (rv[tid + s] == rv[tid] && ri[tid + s] < ri[tid])));
        if (take) { rv[tid] = rv[tid + s]; ri[tid] = ri[tid + s]; }
      }
      __syncthreads();
    }
    if (tid == 0) { idx[(long long)b * k + t] = ri[0]; sv[ri[0]] = -3.4e38f; }
    __syncthreads();
  }
}

// Xs[b,t,h] = X[b, idx[b,t], h] * score[b, idx[b,t]]
__global__ void gather_x_kernel(const float* __restrict__ X, const float* __restrict__ score,
                                const int* __restrict__ idx, float* __restrict__ Xs,
                                int n, int k, long long total) {
  long long i = (long long)blockIdx.x * 256 + threadIdx.x;
  if (i >= total) return;
  int h = (int)(i % H_);
  long long t2 = i / H_;
  int t = (int)(t2 % k);
  int b = (int)(t2 / k);
  int src = idx[(long long)b * k + t];
  Xs[i] = X[((long long)b * n + src) * H_ + h] * score[(long long)b * n + src];
}

// As[b,i,j] = A[b, idx[i], idx[j]]
__global__ void gather_a_kernel(const float* __restrict__ A, const int* __restrict__ idx,
                                float* __restrict__ As, int n, int k, long long total) {
  long long i = (long long)blockIdx.x * 256 + threadIdx.x;
  if (i >= total) return;
  int jj = (int)(i % k);
  long long t2 = i / k;
  int ii = (int)(t2 % k);
  int b  = (int)(t2 / k);
  As[i] = A[(long long)b * n * n + (long long)idx[(long long)b * k + ii] * n + idx[(long long)b * k + jj]];
}

// readout[b, 0:128] += mean_t Xs ; readout[b, 128:256] += max_t Xs
__global__ void readout_kernel(const float* __restrict__ Xs, float* __restrict__ ro,
                               int k, int total) {
  int i = blockIdx.x * 256 + threadIdx.x;
  if (i >= total) return;
  int b = i / H_, h = i % H_;
  const float* xp = Xs + (long long)b * k * H_ + h;
  float s = 0.0f, m = -3.4e38f;
  for (int t = 0; t < k; ++t) { float v = xp[(long long)t * H_]; s += v; m = fmaxf(m, v); }
  ro[(long long)b * 256 + h]       += s / (float)k;
  ro[(long long)b * 256 + 128 + h] += m;
}

// y[b,branch,co] = relu( max_t ( bias + sum_j Z[b, t+j, base+co*kw+j] ) )
__global__ void conv_reduce_kernel(const float* __restrict__ Z,
                                   const float* __restrict__ cb0, const float* __restrict__ cb1,
                                   const float* __restrict__ cb2, const float* __restrict__ cb3,
                                   float* __restrict__ seq, int bstart, int nb) {
  int i = blockIdx.x * 256 + threadIdx.x;
  if (i >= nb * 128) return;
  int c = i % 128, bl = i / 128;
  int branch = c / 32, co = c % 32;
  int kw, base; const float* cb;
  if (branch == 0)      { kw = 3; base = 0;   cb = cb0; }
  else if (branch == 1) { kw = 5; base = 96;  cb = cb1; }
  else if (branch == 2) { kw = 7; base = 256; cb = cb2; }
  else                  { kw = 9; base = 480; cb = cb3; }
  int Lout = L_ - kw + 1;
  const float* zp = Z + (long long)bl * L_ * 768 + base + (long long)co * kw;
  float bias = cb[co];
  float m = -3.4e38f;
  for (int t = 0; t < Lout; ++t) {
    float s = bias;
    for (int j = 0; j < kw; ++j) s += zp[(long long)(t + j) * 768 + j];
    m = fmaxf(m, s);
  }
  seq[(long long)(bstart + bl) * 128 + c] = fmaxf(m, 0.0f);
}

// gc = (1-sigmoid(g))*gnn + sigmoid(g)*seq1
__global__ void fuse_kernel(const float* __restrict__ gnn, const float* __restrict__ seq1,
                            const float* __restrict__ gate, float* __restrict__ gc, int n) {
  int i = blockIdx.x * 256 + threadIdx.x;
  if (i >= n) return;
  float w = 1.0f / (1.0f + expf(-gate[0]));
  gc[i] = (1.0f - w) * gnn[i] + w * seq1[i];
}

__global__ void softmax2_kernel(const float* __restrict__ logits, float* __restrict__ out) {
  int i = threadIdx.x;  // 64 rows
  float a = logits[2 * i], b = logits[2 * i + 1];
  float m = fmaxf(a, b);
  float e0 = expf(a - m), e1 = expf(b - m);
  float s = e0 + e1;
  out[2 * i]     = e0 / s;
  out[2 * i + 1] = e1 / s;
}

static inline unsigned int NB(long long n) { return (unsigned int)((n + 255) / 256); }

extern "C" void kernel_launch(void* const* d_in, const int* in_sizes, int n_in,
                              void* d_out, int out_size, void* d_ws, size_t ws_size,
                              hipStream_t stream) {
  (void)in_sizes; (void)n_in; (void)out_size;

  const float* adj   = (const float*)d_in[0];
  const float* feat  = (const float*)d_in[1];
  const float* pad   = (const float*)d_in[2];
  const float* W1 = (const float*)d_in[3];  const float* b1 = (const float*)d_in[4];  const float* p1 = (const float*)d_in[5];
  const float* W2 = (const float*)d_in[6];  const float* b2 = (const float*)d_in[7];  const float* p2 = (const float*)d_in[8];
  const float* W3 = (const float*)d_in[9];  const float* b3 = (const float*)d_in[10]; const float* p3 = (const float*)d_in[11];
  const float* cat_w = (const float*)d_in[12]; const float* cat_b = (const float*)d_in[13];
  const float* cw0 = (const float*)d_in[14]; const float* cb0 = (const float*)d_in[15];
  const float* cw1 = (const float*)d_in[16]; const float* cb1 = (const float*)d_in[17];
  const float* cw2 = (const float*)d_in[18]; const float* cb2 = (const float*)d_in[19];
  const float* cw3 = (const float*)d_in[20]; const float* cb3 = (const float*)d_in[21];
  const float* tf_w = (const float*)d_in[22]; const float* tf_b = (const float*)d_in[23];
  const float* gate = (const float*)d_in[24];
  const float* fc1_w = (const float*)d_in[25]; const float* fc1_b = (const float*)d_in[26];
  const float* fc2_w = (const float*)d_in[27]; const float* fc2_b = (const float*)d_in[28];
  const float* out_w = (const float*)d_in[29]; const float* out_b = (const float*)d_in[30];

  // ---------------- workspace carve ----------------
  char* base = (char*)d_ws;
  size_t off = 0;
  auto carve = [&](size_t bytes) -> void* {
    off = (off + 255) & ~(size_t)255;
    void* p = base + off;
    off += bytes;
    return p;
  };

  float* dvec = (float*)carve((size_t)B_ * N_ * 4);              // deg^-1/2 (reused per stage)
  float* g    = (float*)carve((size_t)B_ * N_ * H_ * 4);         // GCN output (f32)
  unsigned short* hT = (unsigned short*)carve((size_t)B_ * H_ * N_ * 2);  // h^T bf16 (GEMM CT output)
  float* score = (float*)carve((size_t)B_ * N_ * 4);
  int*   idx   = (int*)carve((size_t)B_ * K1_ * 4);
  float* A1 = (float*)carve((size_t)B_ * K1_ * K1_ * 4);
  float* x1 = (float*)carve((size_t)B_ * K1_ * H_ * 4);
  float* A2 = (float*)carve((size_t)B_ * K2_ * K2_ * 4);
  float* x2 = (float*)carve((size_t)B_ * K2_ * H_ * 4);
  float* x3 = (float*)carve((size_t)B_ * K3_ * H_ * 4);
  float* readout = (float*)carve((size_t)B_ * 256 * 4);
  float* gnn = (float*)carve((size_t)B_ * H_ * 4);
  unsigned short* W1T  = (unsigned short*)carve((size_t)128 * 1024 * 2);
  unsigned short* W2T  = (unsigned short*)carve((size_t)128 * 128 * 2);
  unsigned short* W3T  = (unsigned short*)carve((size_t)128 * 128 * 2);
  unsigned short* catT = (unsigned short*)carve((size_t)128 * 256 * 2);
  unsigned short* tfT  = (unsigned short*)carve((size_t)128 * 128 * 2);
  unsigned short* fc1T = (unsigned short*)carve((size_t)512 * 128 * 2);
  unsigned short* fc2T = (unsigned short*)carve((size_t)256 * 512 * 2);
  unsigned short* outT = (unsigned short*)carve((size_t)128 * 256 * 2);  // N=2 padded to 128
  unsigned short* convT = (unsigned short*)carve((size_t)768 * 1024 * 2);
  const int CB = 8;                                                      // conv batch chunk
  unsigned short* pad16 = (unsigned short*)carve((size_t)CB * L_ * FIN_ * 2); // bf16 A slab
  float* Zslab = (float*)carve((size_t)CB * L_ * 768 * 4);
  float* seq   = (float*)carve((size_t)B_ * 128 * 4);
  float* seq1  = (float*)carve((size_t)B_ * 128 * 4);
  float* gc    = (float*)carve((size_t)B_ * 128 * 4);
  float* f1    = (float*)carve((size_t)B_ * 512 * 4);
  float* f2    = (float*)carve((size_t)B_ * 256 * 4);
  float* logits = (float*)carve((size_t)B_ * 2 * 4);
  if (off > ws_size) return;  // not enough scratch; bail deterministically

  // ---------------- weight packing (bf16, transposed, N padded to 128) ----------------
  packT_kernel<<<NB(128LL * 1024), 256, 0, stream>>>(W1,   W1T, 1024, 128, 128, 128LL * 1024);
  packT_kernel<<<NB(128LL * 128),  256, 0, stream>>>(W2,   W2T,  128, 128, 128, 128LL * 128);
  packT_kernel<<<NB(128LL * 128),  256, 0, stream>>>(W3,   W3T,  128, 128, 128, 128LL * 128);
  packT_kernel<<<NB(128LL * 256),  256, 0, stream>>>(cat_w, catT, 256, 128, 128, 128LL * 256);
  packT_kernel<<<NB(128LL * 128),  256, 0, stream>>>(tf_w, tfT,  128, 128, 128, 128LL * 128);
  packT_kernel<<<NB(512LL * 128),  256, 0, stream>>>(fc1_w, fc1T, 128, 512, 512, 512LL * 128);
  packT_kernel<<<NB(256LL * 512),  256, 0, stream>>>(fc2_w, fc2T, 512, 256, 256, 256LL * 512);
  packT_kernel<<<NB(128LL * 256),  256, 0, stream>>>(out_w, outT, 256,   2, 128, 128LL * 256);
  pack_convw_kernel<<<NB(768LL * 1024), 256, 0, stream>>>(cw0, cw1, cw2, cw3, convT);

  fill_kernel<<<NB(B_ * 256), 256, 0, stream>>>(readout, 0.0f, B_ * 256);

  // ---------------- GCN + pool stage ----------------
  auto gcn_stage = [&](const float* Aadj, const float* X, int n, int F,
                       const unsigned short* WT, const float* bb, const float* pp,
                       int kk, float* Aout, float* Xout) {
    long long tot = (long long)B_ * n;
    deg_kernel<<<NB(tot), 256, 0, stream>>>(Aadj, dvec, n, (int)tot);
    // hT[z] = ((X[z] @ W) * d)^T   (batched, bf16 transposed output feeds next GEMM's B)
    gemm_wmma<0><<<dim3(1, (unsigned)(n / 16), B_), 32, 0, stream>>>(
        X, WT, nullptr, hT, n, H_, F,
        (long long)n * F, 0, (long long)H_ * n, dvec, nullptr, 0);
    // g = relu( (A @ h) * d + b )   (batched, f32 output)
    gemm_wmma<0><<<dim3(1, (unsigned)(n / 16), B_), 32, 0, stream>>>(
        Aadj, hT, g, nullptr, n, H_, n,
        (long long)n * n, (long long)H_ * n, (long long)n * H_, dvec, bb, 1);
    // pooling
    score_kernel<<<NB(tot), 256, 0, stream>>>(g, pp, score, (int)tot);
    topk_kernel<<<B_, 256, 0, stream>>>(score, idx, n, kk);
    long long xt = (long long)B_ * kk * H_;
    gather_x_kernel<<<NB(xt), 256, 0, stream>>>(g, score, idx, Xout, n, kk, xt);
    if (Aout) {
      long long at = (long long)B_ * kk * kk;
      gather_a_kernel<<<NB(at), 256, 0, stream>>>(Aadj, idx, Aout, n, kk, at);
    }
    readout_kernel<<<NB(B_ * H_), 256, 0, stream>>>(Xout, readout, kk, B_ * H_);
  };

  gcn_stage(adj, feat, N_,  FIN_, W1T, b1, p1, K1_, A1, x1);
  gcn_stage(A1,  x1,   K1_, H_,   W2T, b2, p2, K2_, A2, x2);
  gcn_stage(A2,  x2,   K2_, H_,   W3T, b3, p3, K3_, nullptr, x3);  // A3 is never consumed

  // gnn = relu(readout @ cat_w + cat_b)   [64,256]@[256,128]
  gemm_wmma<0><<<dim3(1, 4, 1), 32, 0, stream>>>(
      readout, catT, gnn, nullptr, 64, 128, 256, 0, 0, 0, nullptr, cat_b, 1);

  // ---------------- text-CNN branch: Z = pad @ Wconv, shifted-sum + max ----------------
  for (int ch = 0; ch < B_ / CB; ++ch) {
    long long nel = (long long)CB * L_ * FIN_;
    cvt_bf16_kernel<<<NB(nel), 256, 0, stream>>>(pad + (long long)ch * nel, pad16, nel);
    gemm_wmma<1><<<dim3(6, (unsigned)(CB * L_ / 16), 1), 32, 0, stream>>>(
        pad16, convT, Zslab, nullptr, CB * L_, 768, 1024, 0, 0, 0, nullptr, nullptr, 0);
    conv_reduce_kernel<<<NB(CB * 128), 256, 0, stream>>>(
        Zslab, cb0, cb1, cb2, cb3, seq, ch * CB, CB);
  }
  // seq1 = relu(seq @ tf_w + tf_b)
  gemm_wmma<0><<<dim3(1, 4, 1), 32, 0, stream>>>(
      seq, tfT, seq1, nullptr, 64, 128, 128, 0, 0, 0, nullptr, tf_b, 1);

  // ---------------- gated fusion + MLP head ----------------
  fuse_kernel<<<NB(B_ * 128), 256, 0, stream>>>(gnn, seq1, gate, gc, B_ * 128);
  gemm_wmma<0><<<dim3(4, 4, 1), 32, 0, stream>>>(
      gc, fc1T, f1, nullptr, 64, 512, 128, 0, 0, 0, nullptr, fc1_b, 1);
  gemm_wmma<0><<<dim3(2, 4, 1), 32, 0, stream>>>(
      f1, fc2T, f2, nullptr, 64, 256, 512, 0, 0, 0, nullptr, fc2_b, 1);
  gemm_wmma<0><<<dim3(1, 4, 1), 32, 0, stream>>>(
      f2, outT, logits, nullptr, 64, 2, 256, 0, 0, 0, nullptr, out_b, 1);
  softmax2_kernel<<<1, 64, 0, stream>>>(logits, (float*)d_out);
}
